// TransformerBlock_7619271983784
// MI455X (gfx1250) — matrix operand order repaired and checked by reading, not compiled, not run on hardware
//
#include <hip/hip_runtime.h>
#include <hip/hip_bf16.h>
#include <math.h>

typedef __attribute__((ext_vector_type(16))) __bf16          v16bf;
typedef __attribute__((ext_vector_type(16))) unsigned short  v16u;
typedef __attribute__((ext_vector_type(8)))  float           v8f;

#define HW    16384
#define WIDTH 128
#define C_IN  192
#define NHEAD 8
#define CH    24   // channels per head

static __device__ __forceinline__ unsigned short f2bf(float f) {
    unsigned int u = __float_as_uint(f);
    u = (u + 0x7FFFu + ((u >> 16) & 1u)) >> 16;
    return (unsigned short)u;
}

// ---------------------------------------------------------------------------
// Weight convert fp32 -> bf16 with K padding:  dst[M][Kp], src[M][K]
// ---------------------------------------------------------------------------
__global__ void wconv_kernel(const float* __restrict__ src, unsigned short* __restrict__ dst,
                             int M, int K, int Kp) {
    int i = blockIdx.x * 256 + threadIdx.x;
    if (i >= M * Kp) return;
    int m = i / Kp, k = i % Kp;
    dst[i] = (k < K) ? f2bf(src[m * K + k]) : (unsigned short)0;
}

__global__ void zero_kernel(float* p, int n) {
    int i = blockIdx.x * 256 + threadIdx.x;
    if (i < n) p[i] = 0.f;
}

// ---------------------------------------------------------------------------
// Channel LayerNorm over C=192 per pixel, fp32 in -> bf16 out  [C][HW]
// ---------------------------------------------------------------------------
__global__ __launch_bounds__(256) void ln_kernel(const float* __restrict__ x,
                                                 const float* __restrict__ w,
                                                 const float* __restrict__ b,
                                                 unsigned short* __restrict__ out) {
    int n = blockIdx.x * 256 + threadIdx.x;   // pixel index, grid = HW/256
    float s = 0.f, ss = 0.f;
    for (int c = 0; c < C_IN; ++c) {
        float v = x[c * HW + n];
        s += v; ss += v * v;
    }
    float mu  = s * (1.f / C_IN);
    float var = ss * (1.f / C_IN) - mu * mu;
    float inv = rsqrtf(var + 1e-5f);
    for (int c = 0; c < C_IN; ++c) {
        float v = (x[c * HW + n] - mu) * inv * w[c] + b[c];
        out[c * HW + n] = f2bf(v);
    }
}

// ---------------------------------------------------------------------------
// bf16 WMMA GEMM:  D[M][HW] (f32) = W[M][Kp] (bf16) * X[Kp][HW] (bf16)
// macro tile 128(M) x 128(N); 8 waves arranged 4x2; per-wave 32x64
// (2 A-fragments x 4 B-fragments -> 8 v_wmma per K-step per wave)
// global_prefetch of the next K-tile overlaps with WMMA compute.
// ---------------------------------------------------------------------------
__global__ __launch_bounds__(256) void gemm_bf16_kernel(const unsigned short* __restrict__ W,
                                                        const unsigned short* __restrict__ X,
                                                        float* __restrict__ D,
                                                        int Mreal, int Kp) {
    __shared__ unsigned short As[128 * 32];   // [m][k]  8KB
    __shared__ unsigned short BsT[128 * 32];  // [n][k]  8KB (transposed stage)

    const int t    = threadIdx.x;
    const int lane = t & 31;
    const int wave = t >> 5;
    const int wm   = wave & 3;   // 0..3 -> M offset (32 each)
    const int wn   = wave >> 2;  // 0..1 -> N offset (64 each)
    const int m0   = blockIdx.x * 128;
    const int n0   = blockIdx.y * 128;

    v8f acc[2][4];
    {
        v8f zero = {0.f, 0.f, 0.f, 0.f, 0.f, 0.f, 0.f, 0.f};
        #pragma unroll
        for (int i = 0; i < 2; ++i)
            #pragma unroll
            for (int j = 0; j < 4; ++j) acc[i][j] = zero;
    }

    for (int k0 = 0; k0 < Kp; k0 += 32) {
        // --- stage A tile: 128x32 bf16 = 512 uint4, 2 per thread ---
        #pragma unroll
        for (int v2 = 0; v2 < 2; ++v2) {
            int idx = t + v2 * 256;
            int m = idx >> 2, seg = idx & 3;
            uint4 val = make_uint4(0u, 0u, 0u, 0u);
            if (m0 + m < Mreal)
                val = *(const uint4*)(W + (size_t)(m0 + m) * Kp + k0 + seg * 8);
            *(uint4*)(As + m * 32 + seg * 8) = val;
        }
        // --- stage B tile transposed: read X[k][n0..n0+127], write BsT[n][k] ---
        #pragma unroll
        for (int v2 = 0; v2 < 2; ++v2) {
            int idx = t + v2 * 256;
            int kk = idx >> 4, seg = idx & 15;
            uint4 val = *(const uint4*)(X + (size_t)(k0 + kk) * HW + n0 + seg * 8);
            const unsigned short* e = (const unsigned short*)&val;
            int nl = seg * 8;
            #pragma unroll
            for (int j = 0; j < 8; ++j) BsT[(nl + j) * 32 + kk] = e[j];
        }
        // --- prefetch next K-tile into cache (global_prefetch_b8) ---
        if (k0 + 32 < Kp) {
            if (t < 128) {
                if (m0 + t < Mreal)
                    __builtin_prefetch(W + (size_t)(m0 + t) * Kp + k0 + 32, 0, 1);
            }
            {
                int kk = t >> 3, seg = t & 7;
                __builtin_prefetch(X + (size_t)(k0 + 32 + kk) * HW + n0 + seg * 16, 0, 1);
            }
        }
        __syncthreads();

        #pragma unroll
        for (int mi = 0; mi < 2; ++mi) {
            // A fragment (16x32 bf16): lane -> row (lane&15), K-half by lane group
            int mrow = wm * 32 + mi * 16 + (lane & 15);
            int kb   = (lane < 16) ? 0 : 16;
            const unsigned short* ap = As + mrow * 32;
            v16u au;
            #pragma unroll
            for (int i = 0; i < 8; ++i) au[i] = ap[kb + i];
            #pragma unroll
            for (int i = 0; i < 8; ++i) au[8 + i] = ap[kb + 8 + i];
            v16bf a = __builtin_bit_cast(v16bf, au);

            #pragma unroll
            for (int ni = 0; ni < 4; ++ni) {
                // B fragment (32x16 bf16): lane -> col (lane&15), K-half by group
                int ncol = wn * 64 + ni * 16 + (lane & 15);
                int kb2  = (lane < 16) ? 0 : 16;
                const unsigned short* bp = BsT + ncol * 32 + kb2;
                v16u bu;
                #pragma unroll
                for (int i = 0; i < 16; ++i) bu[i] = bp[i];
                v16bf bbf = __builtin_bit_cast(v16bf, bu);

                acc[mi][ni] = __builtin_amdgcn_wmma_f32_16x16x32_bf16(
                    false, a, false, bbf, (short)0, acc[mi][ni], false, false);
            }
        }
        __syncthreads();
    }

    // --- store D; C/D layout: VGPR r -> M = r + (lane<16?0:8), N = lane&15 ---
    #pragma unroll
    for (int mi = 0; mi < 2; ++mi)
        #pragma unroll
        for (int ni = 0; ni < 4; ++ni) {
            int n     = n0 + wn * 64 + ni * 16 + (lane & 15);
            int mbase = m0 + wm * 32 + mi * 16 + ((lane < 16) ? 0 : 8);
            #pragma unroll
            for (int r = 0; r < 8; ++r) {
                int m = mbase + r;
                if (m < Mreal) D[(size_t)m * HW + n] = acc[mi][ni][r];
            }
        }
}

// ---------------------------------------------------------------------------
// Depthwise 3x3, pad 1, fp32:  in/out [Ch][HW], w [Ch][9]
// ---------------------------------------------------------------------------
__global__ __launch_bounds__(256) void dwconv_kernel(const float* __restrict__ in,
                                                     const float* __restrict__ w9,
                                                     float* __restrict__ out) {
    int idx = blockIdx.x * 256 + threadIdx.x;
    int n = idx & (HW - 1);
    int c = idx >> 14;
    int px = n & (WIDTH - 1), py = n >> 7;
    const float* wp = w9 + c * 9;
    const float* ip = in + (size_t)c * HW;
    float acc = 0.f;
    #pragma unroll
    for (int dy = -1; dy <= 1; ++dy) {
        int yy = py + dy;
        if (yy < 0 || yy >= WIDTH) continue;
        #pragma unroll
        for (int dx = -1; dx <= 1; ++dx) {
            int xx = px + dx;
            if (xx < 0 || xx >= WIDTH) continue;
            acc += wp[(dy + 1) * 3 + (dx + 1)] * ip[yy * WIDTH + xx];
        }
    }
    out[idx] = acc;
}

// ---------------------------------------------------------------------------
// Attention stage 1: accumulate Sqk[h][24][24], qq[h][24], kk[h][24]
// qkvdw layout: q rows 0..191, k rows 192..383, v rows 384..575, each [HW]
// grid = NHEAD * (HW/256); 256 threads
// ---------------------------------------------------------------------------
__global__ __launch_bounds__(256) void sqk_kernel(const float* __restrict__ qkvdw,
                                                  float* __restrict__ Sqk,
                                                  float* __restrict__ qq,
                                                  float* __restrict__ kk) {
    __shared__ float qs[CH][257];
    __shared__ float ks[CH][257];
    const int t    = threadIdx.x;
    const int head = blockIdx.x >> 6;
    const int n0   = (blockIdx.x & 63) * 256;

    for (int i = t; i < CH * 256; i += 256) {
        int c = i >> 8, nn = i & 255;
        qs[c][nn] = qkvdw[(size_t)(head * CH + c) * HW + n0 + nn];
        ks[c][nn] = qkvdw[(size_t)(C_IN + head * CH + c) * HW + n0 + nn];
    }
    __syncthreads();

    for (int job = t; job < CH * CH + 2 * CH; job += 256) {
        if (job < CH * CH) {
            int c = job / CH, d = job % CH;
            float s = 0.f;
            for (int nn = 0; nn < 256; ++nn) s += qs[c][nn] * ks[d][nn];
            atomicAdd(&Sqk[head * CH * CH + job], s);
        } else if (job < CH * CH + CH) {
            int c = job - CH * CH;
            float s = 0.f;
            for (int nn = 0; nn < 256; ++nn) s += qs[c][nn] * qs[c][nn];
            atomicAdd(&qq[head * CH + c], s);
        } else {
            int c = job - CH * CH - CH;
            float s = 0.f;
            for (int nn = 0; nn < 256; ++nn) s += ks[c][nn] * ks[c][nn];
            atomicAdd(&kk[head * CH + c], s);
        }
    }
}

// ---------------------------------------------------------------------------
// Attention stage 2: scale by 1/(||q|| ||k||) * temperature, row softmax
// grid = NHEAD blocks of 32 threads (row r = thread)
// ---------------------------------------------------------------------------
__global__ void softmax_kernel(const float* __restrict__ Sqk,
                               const float* __restrict__ qq,
                               const float* __restrict__ kk,
                               const float* __restrict__ temp,
                               float* __restrict__ attn) {
    int head = blockIdx.x;
    int r = threadIdx.x;
    if (r >= CH) return;
    float qn = fmaxf(sqrtf(qq[head * CH + r]), 1e-12f);
    float tp = temp[head];
    float row[CH];
    float mx = -3.4e38f;
    for (int d = 0; d < CH; ++d) {
        float kn = fmaxf(sqrtf(kk[head * CH + d]), 1e-12f);
        float a = Sqk[head * CH * CH + r * CH + d] / (qn * kn) * tp;
        row[d] = a;
        mx = fmaxf(mx, a);
    }
    float s = 0.f;
    for (int d = 0; d < CH; ++d) { row[d] = expf(row[d] - mx); s += row[d]; }
    float inv = 1.f / s;
    for (int d = 0; d < CH; ++d) attn[head * CH * CH + r * CH + d] = row[d] * inv;
}

// ---------------------------------------------------------------------------
// Attention stage 3: out[h*24+c][n] = sum_d attn[h][c][d] * v[h*24+d][n] -> bf16
// grid = NHEAD * (HW/256)
// ---------------------------------------------------------------------------
__global__ __launch_bounds__(256) void apply_kernel(const float* __restrict__ qkvdw,
                                                    const float* __restrict__ attn,
                                                    unsigned short* __restrict__ out) {
    __shared__ float a[CH * CH];
    const int t    = threadIdx.x;
    const int head = blockIdx.x >> 6;
    const int n    = (blockIdx.x & 63) * 256 + t;
    for (int i = t; i < CH * CH; i += 256) a[i] = attn[head * CH * CH + i];
    __syncthreads();
    float vv[CH];
    #pragma unroll
    for (int d = 0; d < CH; ++d)
        vv[d] = qkvdw[(size_t)(2 * C_IN + head * CH + d) * HW + n];
    for (int c = 0; c < CH; ++c) {
        float acc = 0.f;
        #pragma unroll
        for (int d = 0; d < CH; ++d) acc += a[c * CH + d] * vv[d];
        out[(size_t)(head * CH + c) * HW + n] = f2bf(acc);
    }
}

// ---------------------------------------------------------------------------
// Residual:  out = x + p   (fp32, 192*HW elems)
// ---------------------------------------------------------------------------
__global__ void add_res_kernel(const float* __restrict__ x, const float* __restrict__ p,
                               float* __restrict__ out) {
    int i = blockIdx.x * 256 + threadIdx.x;
    out[i] = x[i] + p[i];
}
__global__ void add_inplace_kernel(float* __restrict__ out, const float* __restrict__ p) {
    int i = blockIdx.x * 256 + threadIdx.x;
    out[i] += p[i];
}

// ---------------------------------------------------------------------------
// Fused FFN depthwise 3x3 + exact-GELU gate -> bf16 [512][HW] (rows 510,511 = 0)
// in: ffn conv out [1020][HW];  w9: [1020][9]
// ---------------------------------------------------------------------------
__global__ __launch_bounds__(256) void gate_kernel(const float* __restrict__ in,
                                                   const float* __restrict__ w9,
                                                   unsigned short* __restrict__ out) {
    int idx = blockIdx.x * 256 + threadIdx.x;   // 512*HW threads
    int n = idx & (HW - 1);
    int c = idx >> 14;
    if (c >= 510) { out[idx] = 0; return; }
    int px = n & (WIDTH - 1), py = n >> 7;
    float d1 = 0.f, d2 = 0.f;
    const float* w1 = w9 + c * 9;
    const float* w2 = w9 + (c + 510) * 9;
    const float* i1 = in + (size_t)c * HW;
    const float* i2 = in + (size_t)(c + 510) * HW;
    #pragma unroll
    for (int dy = -1; dy <= 1; ++dy) {
        int yy = py + dy;
        if (yy < 0 || yy >= WIDTH) continue;
        #pragma unroll
        for (int dx = -1; dx <= 1; ++dx) {
            int xx = px + dx;
            if (xx < 0 || xx >= WIDTH) continue;
            int o = yy * WIDTH + xx, wi = (dy + 1) * 3 + (dx + 1);
            d1 += w1[wi] * i1[o];
            d2 += w2[wi] * i2[o];
        }
    }
    float g = 0.5f * d1 * (1.f + erff(d1 * 0.70710678118654752f)) * d2;
    out[idx] = f2bf(g);
}

// ===========================================================================
extern "C" void kernel_launch(void* const* d_in, const int* in_sizes, int n_in,
                              void* d_out, int out_size, void* d_ws, size_t ws_size,
                              hipStream_t stream) {
    const float* x       = (const float*)d_in[0];
    const float* ln2w    = (const float*)d_in[1];
    const float* ln2b    = (const float*)d_in[2];
    const float* qkvw    = (const float*)d_in[3];
    const float* qkvdww  = (const float*)d_in[4];
    const float* temp    = (const float*)d_in[5];
    const float* projw   = (const float*)d_in[6];
    const float* ln3w    = (const float*)d_in[7];
    const float* ln3b    = (const float*)d_in[8];
    const float* ffn1w   = (const float*)d_in[9];   // [1020][192]
    const float* ffndww  = (const float*)d_in[10];  // [1020][9]
    const float* ffn2w   = (const float*)d_in[11];  // [192][510]

    const int HIDDEN = 510, HID2 = 1020, HIDP = 512;

    // ---- carve workspace ----
    char* p = (char*)d_ws;
    auto carve = [&](size_t bytes) -> void* {
        void* r = (void*)p;
        p += (bytes + 255) & ~(size_t)255;
        return r;
    };
    unsigned short* wqkv_bf  = (unsigned short*)carve((size_t)3 * C_IN * C_IN * 2);
    unsigned short* wproj_bf = (unsigned short*)carve((size_t)C_IN * C_IN * 2);
    unsigned short* wffn1_bf = (unsigned short*)carve((size_t)HID2 * C_IN * 2);
    unsigned short* wffn2_bf = (unsigned short*)carve((size_t)C_IN * HIDP * 2);
    unsigned short* xn_bf    = (unsigned short*)carve((size_t)C_IN * HW * 2);
    unsigned short* ao_bf    = (unsigned short*)carve((size_t)C_IN * HW * 2);
    unsigned short* yn_bf    = (unsigned short*)carve((size_t)C_IN * HW * 2);
    unsigned short* gate_bf  = (unsigned short*)carve((size_t)HIDP * HW * 2);
    float* qkv_f    = (float*)carve((size_t)3 * C_IN * HW * 4);
    float* qkvdw_f  = (float*)carve((size_t)3 * C_IN * HW * 4);
    float* ffn_f    = (float*)carve((size_t)HID2 * HW * 4);
    float* proj_f   = (float*)carve((size_t)C_IN * HW * 4);
    float* ffnout_f = (float*)carve((size_t)C_IN * HW * 4);
    float* Sqk  = (float*)carve((size_t)NHEAD * CH * CH * 4);
    float* qqb  = (float*)carve((size_t)NHEAD * CH * 4);
    float* kkb  = (float*)carve((size_t)NHEAD * CH * 4);
    float* attn = (float*)carve((size_t)NHEAD * CH * CH * 4);

    // ---- weight conversions (once per launch) ----
    wconv_kernel<<<(3 * C_IN * C_IN + 255) / 256, 256, 0, stream>>>(qkvw,  wqkv_bf,  3 * C_IN, C_IN, C_IN);
    wconv_kernel<<<(C_IN * C_IN + 255) / 256, 256, 0, stream>>>(projw, wproj_bf, C_IN, C_IN, C_IN);
    wconv_kernel<<<(HID2 * C_IN + 255) / 256, 256, 0, stream>>>(ffn1w, wffn1_bf, HID2, C_IN, C_IN);
    wconv_kernel<<<(C_IN * HIDP + 255) / 256, 256, 0, stream>>>(ffn2w, wffn2_bf, C_IN, HIDDEN, HIDP);

    const int NT = HW / 128;   // GEMM N tiles = 128

    for (int b = 0; b < 4; ++b) {
        const float* xb   = x + (size_t)b * C_IN * HW;
        float*       outb = (float*)d_out + (size_t)b * C_IN * HW;

        // --- attention branch ---
        ln_kernel<<<HW / 256, 256, 0, stream>>>(xb, ln2w, ln2b, xn_bf);
        gemm_bf16_kernel<<<dim3(5, NT), 256, 0, stream>>>(wqkv_bf, xn_bf, qkv_f, 3 * C_IN, C_IN);
        dwconv_kernel<<<3 * C_IN * HW / 256, 256, 0, stream>>>(qkv_f, qkvdww, qkvdw_f);

        zero_kernel<<<(NHEAD * CH * CH + 255) / 256, 256, 0, stream>>>(Sqk, NHEAD * CH * CH);
        zero_kernel<<<1, 256, 0, stream>>>(qqb, NHEAD * CH);
        zero_kernel<<<1, 256, 0, stream>>>(kkb, NHEAD * CH);
        sqk_kernel<<<NHEAD * (HW / 256), 256, 0, stream>>>(qkvdw_f, Sqk, qqb, kkb);
        softmax_kernel<<<NHEAD, 32, 0, stream>>>(Sqk, qqb, kkb, temp, attn);
        apply_kernel<<<NHEAD * (HW / 256), 256, 0, stream>>>(qkvdw_f, attn, ao_bf);

        gemm_bf16_kernel<<<dim3(2, NT), 256, 0, stream>>>(wproj_bf, ao_bf, proj_f, C_IN, C_IN);
        add_res_kernel<<<C_IN * HW / 256, 256, 0, stream>>>(xb, proj_f, outb);

        // --- FFN branch ---
        ln_kernel<<<HW / 256, 256, 0, stream>>>(outb, ln3w, ln3b, yn_bf);
        gemm_bf16_kernel<<<dim3(8, NT), 256, 0, stream>>>(wffn1_bf, yn_bf, ffn_f, HID2, C_IN);
        gate_kernel<<<HIDP * HW / 256, 256, 0, stream>>>(ffn_f, ffndww, gate_bf);
        gemm_bf16_kernel<<<dim3(2, NT), 256, 0, stream>>>(wffn2_bf, gate_bf, ffnout_f, C_IN, HIDP);
        add_inplace_kernel<<<C_IN * HW / 256, 256, 0, stream>>>(outb, ffnout_f);
    }
}
